// AGAFlashAttention_89558658056829
// MI455X (gfx1250) — matrix operand order repaired and checked by reading, not compiled, not run on hardware
//
#include <hip/hip_runtime.h>
#include <hip/hip_bf16.h>
#include <stdint.h>

// Problem constants (from reference)
#define NB    4
#define NS    4096
#define NTOK  (NB * NS)      // 16384 tokens
#define DDIM  128
#define NSLOT 4096
#define HDIM  1024
#define TOPK  8
#define RSCALE 0.088388347648318447f  // 1/sqrt(128)
#define REPS   1e-10f

typedef __attribute__((ext_vector_type(16))) __bf16        v16bf;
typedef __attribute__((ext_vector_type(8)))  float         v8f;
typedef __attribute__((ext_vector_type(4)))  unsigned int  v4u;
typedef __attribute__((ext_vector_type(4)))  float         v4f;

union BF16x16 { v16bf v; v4u u[2]; };

__device__ __forceinline__ unsigned umax(unsigned a, unsigned b) { return a > b ? a : b; }
__device__ __forceinline__ unsigned umin(unsigned a, unsigned b) { return a < b ? a : b; }
__device__ __forceinline__ v4u ld4u(const void* p) { return *(const v4u*)p; }

// ---------------------------------------------------------------------------
// Kernel 0: convert Q,K to bf16; precompute log(reliability + eps)
// ---------------------------------------------------------------------------
__global__ void aga_prep_kernel(const float* __restrict__ q,
                                const float* __restrict__ k,
                                const float* __restrict__ rel,
                                __bf16* __restrict__ qb,
                                __bf16* __restrict__ kb,
                                float* __restrict__ logrel) {
  int tid = blockIdx.x * blockDim.x + threadIdx.x;
  int stride = gridDim.x * blockDim.x;
  for (int i = tid; i < NTOK * DDIM; i += stride) qb[i] = (__bf16)q[i];
  for (int i = tid; i < NSLOT * DDIM; i += stride) kb[i] = (__bf16)k[i];
  for (int i = tid; i < NSLOT; i += stride) logrel[i] = __logf(rel[i] + REPS);
}

// ---------------------------------------------------------------------------
// Kernel 1: routing GEMM (bf16 WMMA, K·Q^T orientation) with fused top-8.
// One wave handles 32 tokens (two 16-token groups sharing each key tile:
// halves keys L2 traffic, 8 WMMAs per slot tile). D-tile: M = slot rows,
// N = token columns. C layout (32b 16x16): VGPR v -> M=v (lanes 0-15) /
// M=v+8 (lanes 16-31), N = lane&15. Each lane owns ONE token's scores for
// 8 slots per tile: per-lane top-8 list, merged across the lane pair at end.
// Key packing: orderable-fp32 bits, low 12 bits replaced by slot index.
// ---------------------------------------------------------------------------
__device__ __forceinline__ void insert8(unsigned (&m)[8], unsigned c) {
  m[0] = umin(umax(c, m[0]), m[1]);
  m[1] = umin(umax(c, m[1]), m[2]);
  m[2] = umin(umax(c, m[2]), m[3]);
  m[3] = umin(umax(c, m[3]), m[4]);
  m[4] = umin(umax(c, m[4]), m[5]);
  m[5] = umin(umax(c, m[5]), m[6]);
  m[6] = umin(umax(c, m[6]), m[7]);
  m[7] = umax(c, m[7]);
}

#define CE_DESC(a, b) do { unsigned _hi = umax(a, b), _lo = umin(a, b);     \
    (a) = _hi; (b) = _lo; } while (0)

__device__ __forceinline__ void sort8_desc(unsigned (&r)[8]) {
  CE_DESC(r[0], r[1]); CE_DESC(r[2], r[3]); CE_DESC(r[4], r[5]); CE_DESC(r[6], r[7]);
  CE_DESC(r[0], r[2]); CE_DESC(r[1], r[3]); CE_DESC(r[4], r[6]); CE_DESC(r[5], r[7]);
  CE_DESC(r[1], r[2]); CE_DESC(r[5], r[6]);
  CE_DESC(r[0], r[4]); CE_DESC(r[1], r[5]); CE_DESC(r[2], r[6]); CE_DESC(r[3], r[7]);
  CE_DESC(r[2], r[4]); CE_DESC(r[3], r[5]);
  CE_DESC(r[1], r[2]); CE_DESC(r[3], r[4]); CE_DESC(r[5], r[6]);
}

__global__ __launch_bounds__(256)
void aga_router_topk_kernel(const __bf16* __restrict__ qb,
                            const __bf16* __restrict__ kb,
                            const float* __restrict__ logrel,
                            unsigned int* __restrict__ topidx) {
  const int lane = threadIdx.x & 31;
  const int wave = threadIdx.x >> 5;
  const int l    = lane & 15;
  const int hi   = lane >> 4;                  // which lane-half
  const int tokenBase = (blockIdx.x * 8 + wave) * 32;   // 32 tokens per wave

  // B operands (two groups of 16 query tokens), resident across the slot loop.
  // 16-bit B 32x16 as packed here: lanes 0-15 hold K=0..15 of column N=lane,
  // lanes 16-31 hold K=16..31 of column N=lane-16 (contiguous packing).
  BF16x16 bq[2][4];
#pragma unroll
  for (int t = 0; t < 2; ++t) {
    const __bf16* qrow = qb + (size_t)(tokenBase + t * 16 + l) * DDIM;
#pragma unroll
    for (int c = 0; c < 4; ++c) {
      const __bf16* p = qrow + c * 32 + hi * 16;
      bq[t][c].u[0] = ld4u(p);
      bq[t][c].u[1] = ld4u(p + 8);
    }
  }

  // ascending top-8 lists of packed keys, one per token group
  unsigned m[2][8];
#pragma unroll
  for (int t = 0; t < 2; ++t)
#pragma unroll
    for (int i = 0; i < 8; ++i) m[t][i] = 0u;

  for (int st = 0; st < NSLOT; st += 16) {
    // Prefetch the key tile two iterations ahead (gfx1250 global_prefetch_b8).
    if (st + 32 < NSLOT) {
      const __bf16* nk = kb + (size_t)(st + 32 + l) * DDIM;
      __builtin_prefetch(nk, 0, 3);
      __builtin_prefetch((const char*)nk + 128, 0, 3);
    }

    // A operand (key slots). 16-bit A 16x32 as packed here: lane row M = lane&15;
    // lanes 0-15: K = kk+{0..15}; lanes 16-31: K = kk+{16..31}.
    const __bf16* krow = kb + (size_t)(st + l) * DDIM;
    BF16x16 ak[4];
#pragma unroll
    for (int c = 0; c < 4; ++c) {
      const __bf16* p = krow + c * 32 + hi * 16;
      ak[c].u[0] = ld4u(p);
      ak[c].u[1] = ld4u(p + 8);
    }

    v8f acc0 = {0.f, 0.f, 0.f, 0.f, 0.f, 0.f, 0.f, 0.f};
    v8f acc1 = {0.f, 0.f, 0.f, 0.f, 0.f, 0.f, 0.f, 0.f};
#pragma unroll
    for (int c = 0; c < 4; ++c) {
      acc0 = __builtin_amdgcn_wmma_f32_16x16x32_bf16(
          false, ak[c].v, false, bq[0][c].v, (short)0, acc0, false, false);
      acc1 = __builtin_amdgcn_wmma_f32_16x16x32_bf16(
          false, ak[c].v, false, bq[1][c].v, (short)0, acc1, false, false);
    }

    // log-reliability bias for this lane-half's 8 slot rows
    const float* bp = logrel + st + hi * 8;
    v4f b0 = *(const v4f*)bp;
    v4f b1 = *(const v4f*)(bp + 4);
    float bias[8] = {b0.x, b0.y, b0.z, b0.w, b1.x, b1.y, b1.z, b1.w};

#pragma unroll
    for (int t = 0; t < 2; ++t) {
#pragma unroll
      for (int v = 0; v < 8; ++v) {
        float s = (t == 0 ? acc0[v] : acc1[v]) + bias[v];
        unsigned iu = __float_as_uint(s);
        unsigned u  = (iu & 0x80000000u) ? ~iu : (iu | 0x80000000u); // orderable
        unsigned key = (u & 0xFFFFF000u) | (unsigned)(st + v + hi * 8);
        if (key > m[t][0]) insert8(m[t], key);
      }
    }
  }

  // Merge each lane's top-8 with its partner lane (same token, other slots):
  // both lists ascending -> top-8 of union = elementwise max with reversal.
  // Then sort descending so index 0 = best slot, and write indices.
#pragma unroll
  for (int t = 0; t < 2; ++t) {
    unsigned p[8];
#pragma unroll
    for (int i = 0; i < 8; ++i) p[i] = __shfl_xor(m[t][i], 16, 32);
    unsigned r[8];
#pragma unroll
    for (int i = 0; i < 8; ++i) r[i] = umax(m[t][i], p[7 - i]);
    sort8_desc(r);
    if (hi == 0) {
      unsigned* outp = topidx + (size_t)(tokenBase + t * 16 + l) * TOPK;
#pragma unroll
      for (int i = 0; i < 8; ++i) outp[i] = r[i] & 0xFFFu;
    }
  }
}

// ---------------------------------------------------------------------------
// Kernel 2: exact fp32 re-score + softmax + weighted V gather.
// One wave per token (wave32). Lane owns 4 of the 128 q dims for the dots,
// and 32 of the 1024 output dims (8 coalesced float4 chunks).
// ---------------------------------------------------------------------------
__global__ __launch_bounds__(256)
void aga_attend_kernel(const float* __restrict__ q,
                       const float* __restrict__ keys,
                       const float* __restrict__ values,
                       const unsigned int* __restrict__ topidx,
                       float* __restrict__ out,     // [NTOK][HDIM]
                       float* __restrict__ outw) {  // [NTOK][TOPK]
  const int lane  = threadIdx.x & 31;
  const int wave  = threadIdx.x >> 5;
  const int token = blockIdx.x * 8 + wave;

  unsigned idx[TOPK];
#pragma unroll
  for (int j = 0; j < TOPK; ++j) idx[j] = topidx[(size_t)token * TOPK + j];

  // exact dots: d_j = (q . k_idx[j]) * SCALE
  const float* qrow = q + (size_t)token * DDIM;
  v4f q4 = *(const v4f*)(qrow + lane * 4);
  float d[TOPK];
#pragma unroll
  for (int j = 0; j < TOPK; ++j) {
    const float* krow = keys + (size_t)idx[j] * DDIM;
    v4f k4 = *(const v4f*)(krow + lane * 4);
    float p = q4.x * k4.x + q4.y * k4.y + q4.z * k4.z + q4.w * k4.w;
#pragma unroll
    for (int o = 16; o >= 1; o >>= 1) p += __shfl_xor(p, o, 32);
    d[j] = p * RSCALE;
  }

  // stable softmax over 8 (replicated in every lane)
  float mx = d[0];
#pragma unroll
  for (int j = 1; j < TOPK; ++j) mx = fmaxf(mx, d[j]);
  float w[TOPK], sum = 0.f;
#pragma unroll
  for (int j = 0; j < TOPK; ++j) { w[j] = __expf(d[j] - mx); sum += w[j]; }
  float inv = 1.0f / sum;
#pragma unroll
  for (int j = 0; j < TOPK; ++j) w[j] *= inv;

  if (lane == 0) {
    v4f w0 = {w[0], w[1], w[2], w[3]};
    v4f w1 = {w[4], w[5], w[6], w[7]};
    *(v4f*)(outw + (size_t)token * TOPK)     = w0;
    *(v4f*)(outw + (size_t)token * TOPK + 4) = w1;
  }

  // output: out[token][h] = sum_j w_j * V[idx_j][h]
  float* orow = out + (size_t)token * HDIM;
#pragma unroll
  for (int c = 0; c < 8; ++c) {
    int h = c * 128 + lane * 4;
    v4f a = {0.f, 0.f, 0.f, 0.f};
#pragma unroll
    for (int j = 0; j < TOPK; ++j) {
      v4f v4 = *(const v4f*)(values + (size_t)idx[j] * HDIM + h);
      a.x += w[j] * v4.x; a.y += w[j] * v4.y;
      a.z += w[j] * v4.z; a.w += w[j] * v4.w;
    }
    *(v4f*)(orow + h) = a;
  }
}

// ---------------------------------------------------------------------------
extern "C" void kernel_launch(void* const* d_in, const int* in_sizes, int n_in,
                              void* d_out, int out_size, void* d_ws, size_t ws_size,
                              hipStream_t stream) {
  const float* q    = (const float*)d_in[0];  // [4,4096,128]
  const float* keys = (const float*)d_in[1];  // [4096,128]
  const float* vals = (const float*)d_in[2];  // [4096,1024]
  const float* rel  = (const float*)d_in[3];  // [4096]

  float* out  = (float*)d_out;                // [4,4096,1024] flat
  float* outw = out + (size_t)NTOK * HDIM;    // then [4,4096,8] flat

  // workspace layout (~5.5 MB)
  __bf16* qb = (__bf16*)d_ws;                        // 4 MB
  __bf16* kb = qb + (size_t)NTOK * DDIM;             // 1 MB
  float* logrel = (float*)(kb + (size_t)NSLOT * DDIM);  // 16 KB
  unsigned int* topidx = (unsigned int*)(logrel + NSLOT);  // 512 KB

  aga_prep_kernel<<<512, 256, 0, stream>>>(q, keys, rel, qb, kb, logrel);

  // 16384 tokens / (8 waves * 32 tokens per wave) = 64 blocks
  aga_router_topk_kernel<<<64, 256, 0, stream>>>(qb, kb, logrel, topidx);

  // 16384 tokens / 8 waves per block = 2048 blocks
  aga_attend_kernel<<<2048, 256, 0, stream>>>(q, keys, vals, topidx, out, outw);
}
